// EdgeDeltaDiffusion_79517024518477
// MI455X (gfx1250) — matrix operand order repaired and checked by reading, not compiled, not run on hardware
//
#include <hip/hip_runtime.h>
#include <hip/hip_bf16.h>

// Problem constants (match reference)
#define B_  2
#define E_  100000
#define V_  12500
#define D_  128
#define L_  4
#define ROWS 32                         // edges per wave tile
#define TILES (B_ * (E_ / ROWS))        // 6250
#define NTILES 8                        // D/16
#define KS1 8                           // 256/32
#define KS2 4                           // 128/32

typedef __attribute__((ext_vector_type(16))) __bf16 v16bf;
typedef __attribute__((ext_vector_type(8)))  __bf16 v8bf;
typedef __attribute__((ext_vector_type(4)))  __bf16 v4bf;
typedef __attribute__((ext_vector_type(2)))  __bf16 v2bf;
typedef __attribute__((ext_vector_type(8)))  float  v8f;

// f32 -> bf16 (RNE). Prefer the hardware packed convert when the builtin
// exists; fall back to manual integer rounding otherwise.
#if __has_builtin(__builtin_amdgcn_cvt_pk_bf16_f32)
#define HAVE_HW_BF16_CVT 1
#else
#define HAVE_HW_BF16_CVT 0
#endif

__device__ __forceinline__ __bf16 f2bf(float f) {
#if HAVE_HW_BF16_CVT
  v2bf p = __builtin_amdgcn_cvt_pk_bf16_f32(f, 0.0f);
  return p[0];
#else
  unsigned u = __builtin_bit_cast(unsigned, f);
  u = (u + 0x7FFFu + ((u >> 16) & 1u)) >> 16;
  unsigned short s = (unsigned short)u;
  return __builtin_bit_cast(__bf16, s);
#endif
}

__device__ __forceinline__ v4bf f2bf4(float4 v) {
#if HAVE_HW_BF16_CVT
  v2bf lo = __builtin_amdgcn_cvt_pk_bf16_f32(v.x, v.y);
  v2bf hi = __builtin_amdgcn_cvt_pk_bf16_f32(v.z, v.w);
  v4bf r;
  r[0] = lo[0]; r[1] = lo[1]; r[2] = hi[0]; r[3] = hi[1];
  return r;
#else
  v4bf r;
  r[0] = f2bf(v.x); r[1] = f2bf(v.y); r[2] = f2bf(v.z); r[3] = f2bf(v.w);
  return r;
#endif
}

__device__ __forceinline__ float silu(float x) {
  return x / (1.0f + __expf(-x));
}

// ---------------------------------------------------------------------------
// Pack fp32 weights (Ktot x 128, row-major, L layers) into bf16 WMMA B-fragment
// order. Fragment (ks, nt): lane l holds col = nt*16 + (l&15), elements i=0..15
// map to K = ks*32 + (l>>4)*8 + i for i<8, ks*32 + 16 + (l>>4)*8 + (i-8) for i>=8.
// ---------------------------------------------------------------------------
__global__ __launch_bounds__(256) void pack_w_kernel(const float* __restrict__ W,
                                                     __bf16* __restrict__ out,
                                                     int Ktot, int layers) {
  int perLayer = (Ktot / 32) * NTILES * 512;
  int total = layers * perLayer;
  int gid = blockIdx.x * blockDim.x + threadIdx.x;
  if (gid >= total) return;
  int i   = gid / perLayer;
  int rem = gid % perLayer;
  int f   = rem / 512;      // fragment index: ks*8 + nt
  int e   = rem % 512;
  int lane = e / 16;
  int idx  = e % 16;
  int ks = f / NTILES;
  int nt = f % NTILES;
  int col = nt * 16 + (lane & 15);
  int K   = ks * 32 + (lane >> 4) * 8 + (idx < 8 ? idx : 16 + (idx - 8));
  out[gid] = f2bf(W[(size_t)i * Ktot * D_ + (size_t)K * D_ + col]);
}

// ---------------------------------------------------------------------------
// Sinusoidal timestep embedding + 2-layer silu MLP: one block per batch row.
// ---------------------------------------------------------------------------
__global__ __launch_bounds__(128) void time_embed_kernel(const int* __restrict__ n,
                                                         const float* __restrict__ te_w1,
                                                         const float* __restrict__ te_b1,
                                                         const float* __restrict__ te_w2,
                                                         const float* __restrict__ te_b2,
                                                         float* __restrict__ t_emb) {
  __shared__ float emb[D_];
  __shared__ float hid[4 * D_];
  int b = blockIdx.x;
  int t = threadIdx.x;
  float nv = (float)n[b];
  int idx = t & 63;
  float f = __expf(-9.210340371976184f * (float)idx * (1.0f / 64.0f)); // ln(1e4)
  float a = nv * f;
  emb[t] = (t < 64) ? __sinf(a) : __cosf(a);
  __syncthreads();
  for (int j = t; j < 4 * D_; j += 128) {
    float s = te_b1[j];
    for (int k = 0; k < D_; ++k) s += emb[k] * te_w1[k * 4 * D_ + j];
    hid[j] = silu(s);
  }
  __syncthreads();
  float s = te_b2[t];
  for (int k = 0; k < 4 * D_; ++k) s += hid[k] * te_w2[k * D_ + t];
  t_emb[b * D_ + t] = s;
}

// ---------------------------------------------------------------------------
// Input projection: h[b,e,:] = delta[b,e,:] @ w_in + b_in + t_emb[b]. One wave
// per edge; lane handles 4 contiguous features (float4 store).
// ---------------------------------------------------------------------------
__global__ __launch_bounds__(256) void input_proj_kernel(const float* __restrict__ delta,
                                                         const float* __restrict__ w_in,
                                                         const float* __restrict__ b_in,
                                                         const float* __restrict__ t_emb,
                                                         float* __restrict__ h) {
  int gid = blockIdx.x * blockDim.x + threadIdx.x;
  int wave = gid >> 5;
  int lane = gid & 31;
  if (wave >= B_ * E_) return;
  int b = wave / E_;
  const float* dp = delta + (size_t)wave * 3;
  float d0 = dp[0], d1 = dp[1], d2 = dp[2];
  int c = lane * 4;
  float4 o;
  float* op = &o.x;
  #pragma unroll
  for (int k = 0; k < 4; ++k) {
    int cc = c + k;
    op[k] = d0 * w_in[cc] + d1 * w_in[D_ + cc] + d2 * w_in[2 * D_ + cc]
          + b_in[cc] + t_emb[b * D_ + cc];
  }
  *(float4*)(h + (size_t)wave * D_ + c) = o;
}

// ---------------------------------------------------------------------------
// Scatter-add: vf[b, src[e]] += h[b,e]; vf[b, dst[e]] += h[b,e].
// One wave per edge, f32 global atomics (L2-resident: vf = 6.4 MB/batch).
// ---------------------------------------------------------------------------
__global__ __launch_bounds__(256) void scatter_kernel(const float* __restrict__ h,
                                                      const int* __restrict__ edge_index,
                                                      float* __restrict__ vf) {
  int gid = blockIdx.x * blockDim.x + threadIdx.x;
  int wave = gid >> 5;
  int lane = gid & 31;
  if (wave >= B_ * E_) return;
  int b = wave / E_;
  int e = wave - b * E_;
  int s = edge_index[e];
  int d = edge_index[E_ + e];
  float4 v = *(const float4*)(h + (size_t)wave * D_ + lane * 4);
  float* vs = vf + ((size_t)b * V_ + s) * D_ + lane * 4;
  float* vd = vf + ((size_t)b * V_ + d) * D_ + lane * 4;
  const float* vp = &v.x;
  #pragma unroll
  for (int k = 0; k < 4; ++k)
    __hip_atomic_fetch_add(vs + k, vp[k], __ATOMIC_RELAXED, __HIP_MEMORY_SCOPE_AGENT);
  #pragma unroll
  for (int k = 0; k < 4; ++k)
    __hip_atomic_fetch_add(vd + k, vp[k], __ATOMIC_RELAXED, __HIP_MEMORY_SCOPE_AGENT);
}

// ---------------------------------------------------------------------------
// Fused layer kernel: one wave per 32-edge tile (two 16-row WMMA A tiles share
// every B fragment -> 2x weight reuse per vmem op).
//   gather [vf[src] || vf[dst]] -> LDS bf16 (32x256)
//   GEMM1 (32x256)x(256x128) via 8 ksteps x 8 ntiles x 2 row-tiles wmma
//   +b1, LayerNorm (shfl_xor over 16-lane half), silu -> restage LDS bf16
//   GEMM2 (32x128)x(128x128)
//   h = LayerNorm(h + x + b2) with norm_g/norm_b
// No block-wide barriers -> early-return tail guard is safe.
// ---------------------------------------------------------------------------
__global__ __launch_bounds__(128) void layer_kernel(float* __restrict__ h,
                                                    const float* __restrict__ vf,
                                                    const int* __restrict__ edge_index,
                                                    const __bf16* __restrict__ wp1,
                                                    const __bf16* __restrict__ wp2,
                                                    const float* __restrict__ b1,
                                                    const float* __restrict__ lng,
                                                    const float* __restrict__ lnb,
                                                    const float* __restrict__ b2,
                                                    const float* __restrict__ ng,
                                                    const float* __restrict__ nb) {
  __shared__ __align__(16) __bf16 xs[4][ROWS][256];
  const int w    = threadIdx.x >> 5;
  const int l    = threadIdx.x & 31;
  const int tile = blockIdx.x * 4 + w;
  if (tile >= TILES) return;
  const int b    = tile / (E_ / ROWS);
  const int e0   = (tile - b * (E_ / ROWS)) * ROWS;
  const int m    = l & 15;                 // row-within-16 / output column
  const int g    = l >> 4;                 // lane half (rows j vs j+8)
  const int col  = m;

  // ---- stage x = [vf[src] || vf[dst]] into LDS bf16: lane l stages edge l
  {
    #pragma unroll
    for (int part = 0; part < 2; ++part) {
      int vtx = edge_index[part * E_ + e0 + l];
      const float* vrow = vf + ((size_t)b * V_ + vtx) * D_;
      __bf16* xrow = &xs[w][l][part * D_];
      #pragma unroll
      for (int c = 0; c < D_; c += 4) {
        float4 v = *(const float4*)(vrow + c);
        *(v4bf*)(xrow + c) = f2bf4(v);
      }
    }
  }
  asm volatile("s_wait_dscnt 0" ::: "memory");

  const __bf16* xa0 = &xs[w][m][0];        // A rows 0..15
  const __bf16* xa1 = &xs[w][16 + m][0];   // A rows 16..31

  // ---- GEMM1: acc[t][nt], K = 256; each B fragment feeds 2 WMMAs
  v8f acc[2][NTILES];
  #pragma unroll
  for (int t = 0; t < 2; ++t)
    #pragma unroll
    for (int nt = 0; nt < NTILES; ++nt) acc[t][nt] = {};
  #pragma unroll
  for (int ks = 0; ks < KS1; ++ks) {
    const int kb = ks * 32 + g * 8;
    v8bf lo0 = *(const v8bf*)(xa0 + kb);
    v8bf hi0 = *(const v8bf*)(xa0 + kb + 16);
    v8bf lo1 = *(const v8bf*)(xa1 + kb);
    v8bf hi1 = *(const v8bf*)(xa1 + kb + 16);
    v16bf a0 = __builtin_shufflevector(lo0, hi0, 0,1,2,3,4,5,6,7,8,9,10,11,12,13,14,15);
    v16bf a1 = __builtin_shufflevector(lo1, hi1, 0,1,2,3,4,5,6,7,8,9,10,11,12,13,14,15);
    #pragma unroll
    for (int nt = 0; nt < NTILES; ++nt) {
      v16bf bf = *(const v16bf*)(wp1 + ((size_t)(ks * NTILES + nt) * 512) + l * 16);
      acc[0][nt] = __builtin_amdgcn_wmma_f32_16x16x32_bf16(
          false, a0, false, bf, (short)0, acc[0][nt], false, false);
      acc[1][nt] = __builtin_amdgcn_wmma_f32_16x16x32_bf16(
          false, a1, false, bf, (short)0, acc[1][nt], false, false);
    }
  }

  // ---- +b1, LayerNorm(ln_g, ln_b), silu -> restage bf16 into LDS cols 0..127
  {
    float b1v[NTILES], gv[NTILES], bv[NTILES];
    #pragma unroll
    for (int nt = 0; nt < NTILES; ++nt) {
      b1v[nt] = b1[nt * 16 + col];
      gv[nt]  = lng[nt * 16 + col];
      bv[nt]  = lnb[nt * 16 + col];
    }
    #pragma unroll
    for (int t = 0; t < 2; ++t) {
      #pragma unroll
      for (int j = 0; j < 8; ++j) {
        float xv[NTILES];
        float s1 = 0.f, s2 = 0.f;
        #pragma unroll
        for (int nt = 0; nt < NTILES; ++nt) {
          float x = acc[t][nt][j] + b1v[nt];
          xv[nt] = x; s1 += x; s2 += x * x;
        }
        #pragma unroll
        for (int msk = 1; msk <= 8; msk <<= 1) {
          s1 += __shfl_xor(s1, msk, 32);
          s2 += __shfl_xor(s2, msk, 32);
        }
        float mean = s1 * (1.0f / 128.0f);
        float var  = s2 * (1.0f / 128.0f) - mean * mean;
        float rstd = rsqrtf(var + 1e-5f);
        int r = t * 16 + j + g * 8;
        #pragma unroll
        for (int nt = 0; nt < NTILES; ++nt) {
          float y = silu((xv[nt] - mean) * rstd * gv[nt] + bv[nt]);
          xs[w][r][nt * 16 + col] = f2bf(y);
        }
      }
    }
  }
  asm volatile("s_wait_dscnt 0" ::: "memory");

  // ---- GEMM2: K = 128, same 2x B reuse
  v8f acc2[2][NTILES];
  #pragma unroll
  for (int t = 0; t < 2; ++t)
    #pragma unroll
    for (int nt = 0; nt < NTILES; ++nt) acc2[t][nt] = {};
  #pragma unroll
  for (int ks = 0; ks < KS2; ++ks) {
    const int kb = ks * 32 + g * 8;
    v8bf lo0 = *(const v8bf*)(xa0 + kb);
    v8bf hi0 = *(const v8bf*)(xa0 + kb + 16);
    v8bf lo1 = *(const v8bf*)(xa1 + kb);
    v8bf hi1 = *(const v8bf*)(xa1 + kb + 16);
    v16bf a0 = __builtin_shufflevector(lo0, hi0, 0,1,2,3,4,5,6,7,8,9,10,11,12,13,14,15);
    v16bf a1 = __builtin_shufflevector(lo1, hi1, 0,1,2,3,4,5,6,7,8,9,10,11,12,13,14,15);
    #pragma unroll
    for (int nt = 0; nt < NTILES; ++nt) {
      v16bf bf = *(const v16bf*)(wp2 + ((size_t)(ks * NTILES + nt) * 512) + l * 16);
      acc2[0][nt] = __builtin_amdgcn_wmma_f32_16x16x32_bf16(
          false, a0, false, bf, (short)0, acc2[0][nt], false, false);
      acc2[1][nt] = __builtin_amdgcn_wmma_f32_16x16x32_bf16(
          false, a1, false, bf, (short)0, acc2[1][nt], false, false);
    }
  }

  // ---- h = LayerNorm(h + x + b2; norm_g, norm_b)
  {
    float b2v[NTILES], gv[NTILES], bv[NTILES];
    #pragma unroll
    for (int nt = 0; nt < NTILES; ++nt) {
      b2v[nt] = b2[nt * 16 + col];
      gv[nt]  = ng[nt * 16 + col];
      bv[nt]  = nb[nt * 16 + col];
    }
    float* hb = h + ((size_t)b * E_ + e0) * D_;
    #pragma unroll
    for (int t = 0; t < 2; ++t) {
      #pragma unroll
      for (int j = 0; j < 8; ++j) {
        int r = t * 16 + j + g * 8;
        float xv[NTILES];
        float s1 = 0.f, s2 = 0.f;
        #pragma unroll
        for (int nt = 0; nt < NTILES; ++nt) {
          float x = hb[r * D_ + nt * 16 + col] + acc2[t][nt][j] + b2v[nt];
          xv[nt] = x; s1 += x; s2 += x * x;
        }
        #pragma unroll
        for (int msk = 1; msk <= 8; msk <<= 1) {
          s1 += __shfl_xor(s1, msk, 32);
          s2 += __shfl_xor(s2, msk, 32);
        }
        float mean = s1 * (1.0f / 128.0f);
        float var  = s2 * (1.0f / 128.0f) - mean * mean;
        float rstd = rsqrtf(var + 1e-5f);
        #pragma unroll
        for (int nt = 0; nt < NTILES; ++nt)
          hb[r * D_ + nt * 16 + col] = (xv[nt] - mean) * rstd * gv[nt] + bv[nt];
      }
    }
  }
}

// ---------------------------------------------------------------------------
// Output projection: out[b,e,:] = h[b,e,:] @ w_out + b_out. One wave per edge.
// ---------------------------------------------------------------------------
__global__ __launch_bounds__(256) void out_proj_kernel(const float* __restrict__ h,
                                                       const float* __restrict__ w_out,
                                                       const float* __restrict__ b_out,
                                                       float* __restrict__ out) {
  int gid = blockIdx.x * blockDim.x + threadIdx.x;
  int wave = gid >> 5;
  int lane = gid & 31;
  if (wave >= B_ * E_) return;
  float4 v = *(const float4*)(h + (size_t)wave * D_ + lane * 4);
  const float* vp = &v.x;
  float a0 = 0.f, a1 = 0.f, a2 = 0.f;
  #pragma unroll
  for (int k = 0; k < 4; ++k) {
    int c = lane * 4 + k;
    a0 += vp[k] * w_out[c * 3 + 0];
    a1 += vp[k] * w_out[c * 3 + 1];
    a2 += vp[k] * w_out[c * 3 + 2];
  }
  #pragma unroll
  for (int msk = 1; msk <= 16; msk <<= 1) {
    a0 += __shfl_xor(a0, msk, 32);
    a1 += __shfl_xor(a1, msk, 32);
    a2 += __shfl_xor(a2, msk, 32);
  }
  if (lane == 0) {
    float* op = out + (size_t)wave * 3;
    op[0] = a0 + b_out[0];
    op[1] = a1 + b_out[1];
    op[2] = a2 + b_out[2];
  }
}

// ---------------------------------------------------------------------------
extern "C" void kernel_launch(void* const* d_in, const int* in_sizes, int n_in,
                              void* d_out, int out_size, void* d_ws, size_t ws_size,
                              hipStream_t stream) {
  const float* delta   = (const float*)d_in[0];
  const int*   nvec    = (const int*)d_in[1];
  const int*   eidx    = (const int*)d_in[2];
  // d_in[3] = num_vertices (compile-time constant V_)
  const float* w_in    = (const float*)d_in[4];
  const float* b_in    = (const float*)d_in[5];
  const float* W1      = (const float*)d_in[6];
  const float* b1      = (const float*)d_in[7];
  const float* ln_g    = (const float*)d_in[8];
  const float* ln_b    = (const float*)d_in[9];
  const float* W2      = (const float*)d_in[10];
  const float* b2      = (const float*)d_in[11];
  const float* norm_g  = (const float*)d_in[12];
  const float* norm_b  = (const float*)d_in[13];
  const float* te_w1   = (const float*)d_in[14];
  const float* te_b1   = (const float*)d_in[15];
  const float* te_w2   = (const float*)d_in[16];
  const float* te_b2   = (const float*)d_in[17];
  const float* w_out   = (const float*)d_in[18];
  const float* b_out   = (const float*)d_in[19];
  float* out = (float*)d_out;

  char* ws = (char*)d_ws;
  size_t off = 0;
  float* h    = (float*)(ws + off); off += (size_t)B_ * E_ * D_ * sizeof(float);
  float* vf   = (float*)(ws + off); off += (size_t)B_ * V_ * D_ * sizeof(float);
  float* temb = (float*)(ws + off); off += (size_t)B_ * D_ * sizeof(float);
  off = (off + 255) & ~(size_t)255;
  __bf16* wp1 = (__bf16*)(ws + off); off += (size_t)L_ * KS1 * NTILES * 512 * sizeof(__bf16);
  __bf16* wp2 = (__bf16*)(ws + off); off += (size_t)L_ * KS2 * NTILES * 512 * sizeof(__bf16);

  // 1. pack weights into WMMA B-fragment bf16 layout (cheap; deterministic)
  {
    int t1 = L_ * KS1 * NTILES * 512;  // 131072
    int t2 = L_ * KS2 * NTILES * 512;  // 65536
    pack_w_kernel<<<(t1 + 255) / 256, 256, 0, stream>>>(W1, wp1, 2 * D_, L_);
    pack_w_kernel<<<(t2 + 255) / 256, 256, 0, stream>>>(W2, wp2, D_, L_);
  }

  // 2. timestep embedding MLP
  time_embed_kernel<<<B_, 128, 0, stream>>>(nvec, te_w1, te_b1, te_w2, te_b2, temb);

  // 3. input projection + broadcast t_emb
  {
    int waves = B_ * E_;
    input_proj_kernel<<<(waves * 32 + 255) / 256, 256, 0, stream>>>(
        delta, w_in, b_in, temb, h);
  }

  // 4. message-passing layers
  for (int i = 0; i < L_; ++i) {
    hipMemsetAsync(vf, 0, (size_t)B_ * V_ * D_ * sizeof(float), stream);
    int waves = B_ * E_;
    scatter_kernel<<<(waves * 32 + 255) / 256, 256, 0, stream>>>(h, eidx, vf);
    layer_kernel<<<(TILES + 3) / 4, 128, 0, stream>>>(
        h, vf, eidx,
        wp1 + (size_t)i * KS1 * NTILES * 512,
        wp2 + (size_t)i * KS2 * NTILES * 512,
        b1 + i * D_, ln_g + i * D_, ln_b + i * D_,
        b2 + i * D_, norm_g + i * D_, norm_b + i * D_);
  }

  // 5. output projection
  {
    int waves = B_ * E_;
    out_proj_kernel<<<(waves * 32 + 255) / 256, 256, 0, stream>>>(h, w_out, b_out, out);
  }
}
